// DeepFMBackbone_70437463654828
// MI455X (gfx1250) — matrix operand order repaired and checked by reading, not compiled, not run on hardware
//
#include <hip/hip_runtime.h>
#include <hip/hip_bf16.h>

// ---------------------------------------------------------------------------
// DeepFM backbone forward, MI455X (gfx1250, wave32).
// GEMMs use split-precision bf16 (hi+lo) with V_WMMA_F32_16X16X32_BF16,
// f32 accumulate: z = Ahi*Bhi + Ahi*Blo + Alo*Bhi  (~fp32 accuracy).
// ---------------------------------------------------------------------------

typedef __attribute__((ext_vector_type(16))) __bf16 v16bf;
typedef __attribute__((ext_vector_type(8)))  float  v8f;
typedef __attribute__((ext_vector_type(4)))  unsigned int u32x4;

union V16BF { u32x4 q[2]; v16bf v; };

#define B_ROWS 16384
#define KPAD1  544     // 33*16 = 528 padded to multiple of 32
#define N1     256
#define K2     256
#define N2     128

__constant__ int c_offs[20] = {
    0, 1000000, 1100000, 1200000, 1210000, 1220000, 1230000, 1231000,
    1232000, 1233000, 1234000, 1234100, 1234200, 1234300, 1234400,
    1234450, 1234500, 1234520, 1234540, 1234550};

// ---------------------------------------------------------------------------
// K0: one wave per row. Gather sparse embeddings, compute dense embeddings,
// write embed row as bf16 hi/lo (padded to 544 cols), compute fm1+fm2.
// Lane = 16*h + f;  h=0 handles fields 0..16, h=1 handles 17..32 (+pad 33).
// ---------------------------------------------------------------------------
__global__ void k_embed(const float* __restrict__ x, const float* __restrict__ emb,
                        const float* __restrict__ fm1t, const float* __restrict__ dw,
                        const float* __restrict__ db, const float* __restrict__ dfw,
                        const float* __restrict__ dfb, __bf16* __restrict__ A1hi,
                        __bf16* __restrict__ A1lo, float* __restrict__ fmsum) {
  const int lane = threadIdx.x & 31;
  const int row = (int)(((size_t)blockIdx.x * blockDim.x + threadIdx.x) >> 5);
  const int f = lane & 15;
  const int h = lane >> 4;
  const float* xr = x + (size_t)row * 33;
  __bf16* arow_hi = A1hi + (size_t)row * KPAD1;
  __bf16* arow_lo = A1lo + (size_t)row * KPAD1;

  float s = 0.f, ssq = 0.f;
  for (int i = 0; i < 17; ++i) {
    int fld = h ? (17 + i) : i;
    float e = 0.f;
    if (fld < 20) {
      int gid = (int)xr[fld] + c_offs[fld];
      e = emb[(size_t)gid * 16 + f];
    } else if (fld < 33) {
      int j = fld - 20;
      e = xr[fld] * dw[j * 16 + f] + db[j * 16 + f];
    }
    __bf16 hi = (__bf16)e;
    arow_hi[fld * 16 + f] = hi;
    arow_lo[fld * 16 + f] = (__bf16)(e - (float)hi);
    s += e;
    ssq += e * e;
  }
  // combine the two field-halves per feature
  s   += __shfl_xor(s, 16);
  ssq += __shfl_xor(ssq, 16);
  float v = s * s - ssq;
  v += __shfl_xor(v, 1); v += __shfl_xor(v, 2);
  v += __shfl_xor(v, 4); v += __shfl_xor(v, 8);
  float fm2 = 0.5f * v;

  // fm1: lanes 0..19 sparse table, lanes 20..31 dense j=0..11, lane 0 extras
  float c = 0.f;
  if (lane < 20) {
    int gid = (int)xr[lane] + c_offs[lane];
    c = fm1t[gid];
  } else {
    int j = lane - 20;  // 0..11
    c = xr[lane] * dfw[j];
  }
  if (lane == 0) {
    c += xr[32] * dfw[12];
    float bs = 0.f;
    for (int j = 0; j < 13; ++j) bs += dfb[j];
    c += bs;
  }
  c += __shfl_xor(c, 1);  c += __shfl_xor(c, 2);
  c += __shfl_xor(c, 4);  c += __shfl_xor(c, 8);
  c += __shfl_xor(c, 16);
  if (lane == 0) fmsum[row] = c + fm2;
}

// ---------------------------------------------------------------------------
// Convert weight matrix (K x N row-major f32) into transposed bf16 hi/lo
// [N][Kpad] with zero padding beyond K.
// ---------------------------------------------------------------------------
__global__ void k_convW(const float* __restrict__ W, __bf16* __restrict__ hiT,
                        __bf16* __restrict__ loT, int K, int Kpad, int N) {
  int i = blockIdx.x * blockDim.x + threadIdx.x;
  if (i >= N * Kpad) return;
  int n = i / Kpad;
  int k = i - n * Kpad;
  float w = (k < K) ? W[(size_t)k * N + n] : 0.f;
  __bf16 hi = (__bf16)w;
  hiT[(size_t)n * Kpad + k] = hi;
  loT[(size_t)n * Kpad + k] = (__bf16)(w - (float)hi);
}

// ---------------------------------------------------------------------------
// GEMM: Z(MxN) = A(MxK) * B^T(NxK) + bias, split-precision bf16 WMMA.
// One wave computes a 16x64 strip (4 accumulators). The three split-precision
// products are interleaved across the 4 accumulators so consecutive WMMAs are
// independent (no XDL bubbles). Waves within a block share the same 64-column
// strip (same B fragments -> B stays hot in WGP$); consecutive mt per wave.
// ---------------------------------------------------------------------------
__global__ void k_gemm(const __bf16* __restrict__ Ahi, const __bf16* __restrict__ Alo,
                       const __bf16* __restrict__ Bhi, const __bf16* __restrict__ Blo,
                       const float* __restrict__ bias, float* __restrict__ Z,
                       int M, int K, int N) {
  const int lane = threadIdx.x & 31;
  const int gw = (int)(((size_t)blockIdx.x * blockDim.x + threadIdx.x) >> 5);
  const int mTiles = M >> 4;
  const int mt = gw % mTiles;   // consecutive waves -> consecutive row tiles
  const int nb = gw / mTiles;   // block shares the column strip
  if (nb * 64 >= N) return;

  const int mrow = mt * 16 + (lane & 15);
  const int kA = (lane < 16) ? 0 : 16;  // 16-bit A fragment K base per half
  const int kB = (lane < 16) ? 0 : 16;  // 16-bit B fragment K base per half
  const int n0 = nb * 64 + (lane & 15);

  v8f acc[4];
  const v8f zero = {0.f, 0.f, 0.f, 0.f, 0.f, 0.f, 0.f, 0.f};
#pragma unroll
  for (int t = 0; t < 4; ++t) acc[t] = zero;

  const __bf16* arow_hi = Ahi + (size_t)mrow * K;
  const __bf16* arow_lo = Alo + (size_t)mrow * K;

  for (int ks = 0; ks < K; ks += 32) {
    V16BF ah, al;
    ah.q[0] = *(const u32x4*)(arow_hi + ks + kA);
    ah.q[1] = *(const u32x4*)(arow_hi + ks + kA + 8);
    al.q[0] = *(const u32x4*)(arow_lo + ks + kA);
    al.q[1] = *(const u32x4*)(arow_lo + ks + kA + 8);

    V16BF bh[4], bl[4];
#pragma unroll
    for (int t = 0; t < 4; ++t) {
      const __bf16* bh_p = Bhi + (size_t)(n0 + t * 16) * K + ks + kB;
      const __bf16* bl_p = Blo + (size_t)(n0 + t * 16) * K + ks + kB;
      bh[t].q[0] = *(const u32x4*)(bh_p);
      bh[t].q[1] = *(const u32x4*)(bh_p + 8);
      bl[t].q[0] = *(const u32x4*)(bl_p);
      bl[t].q[1] = *(const u32x4*)(bl_p + 8);
    }
    // Interleave products across accumulators: dependency distance 4.
#pragma unroll
    for (int t = 0; t < 4; ++t)
      acc[t] = __builtin_amdgcn_wmma_f32_16x16x32_bf16(false, ah.v, false, bh[t].v,
                                                       (short)0, acc[t], false, false);
#pragma unroll
    for (int t = 0; t < 4; ++t)
      acc[t] = __builtin_amdgcn_wmma_f32_16x16x32_bf16(false, ah.v, false, bl[t].v,
                                                       (short)0, acc[t], false, false);
#pragma unroll
    for (int t = 0; t < 4; ++t)
      acc[t] = __builtin_amdgcn_wmma_f32_16x16x32_bf16(false, al.v, false, bh[t].v,
                                                       (short)0, acc[t], false, false);
  }

  const int r0 = mt * 16 + ((lane < 16) ? 0 : 8);
#pragma unroll
  for (int t = 0; t < 4; ++t) {
    int n = n0 + t * 16;
    float bn = bias[n];
#pragma unroll
    for (int r = 0; r < 8; ++r) {
      Z[(size_t)(r0 + r) * N + n] = acc[t][r] + bn;
    }
  }
}

// ---------------------------------------------------------------------------
// Deterministic batch statistics (two-pass tree): per-column sum / sumsq.
// ---------------------------------------------------------------------------
__global__ void k_stats1(const float* __restrict__ Z, float* __restrict__ ps,
                         float* __restrict__ pss, int N, int rowsPerBlk) {
  int n = threadIdx.x;  // blockDim.x == N
  size_t base = (size_t)blockIdx.x * rowsPerBlk * N + n;
  float s = 0.f, ss = 0.f;
  for (int r = 0; r < rowsPerBlk; ++r) {
    float v = Z[base + (size_t)r * N];
    s += v;
    ss += v * v;
  }
  ps[blockIdx.x * N + n] = s;
  pss[blockIdx.x * N + n] = ss;
}

__global__ void k_stats2(const float* __restrict__ ps, const float* __restrict__ pss,
                         float* __restrict__ mean, float* __restrict__ inv,
                         int N, int nblk, float invM) {
  int n = threadIdx.x;  // blockDim.x == N
  float s = 0.f, ss = 0.f;
  for (int b = 0; b < nblk; ++b) {
    s += ps[b * N + n];
    ss += pss[b * N + n];
  }
  float m = s * invM;
  float var = ss * invM - m * m;
  mean[n] = m;
  inv[n] = rsqrtf(var + 1e-5f);
}

// ---------------------------------------------------------------------------
// BN + ReLU, then split into bf16 hi/lo for the next GEMM's A operand.
// N must be a power of two (here 256).
// ---------------------------------------------------------------------------
__global__ void k_bnsplit(const float* __restrict__ Z, const float* __restrict__ mean,
                          const float* __restrict__ inv, const float* __restrict__ g,
                          const float* __restrict__ be, __bf16* __restrict__ Hhi,
                          __bf16* __restrict__ Hlo, int total, int Nmask) {
  int i = blockIdx.x * blockDim.x + threadIdx.x;
  if (i >= total) return;
  int n = i & Nmask;
  float y = g[n] * (Z[i] - mean[n]) * inv[n] + be[n];
  y = fmaxf(y, 0.f);
  __bf16 hi = (__bf16)y;
  Hhi[i] = hi;
  Hlo[i] = (__bf16)(y - (float)hi);
}

// ---------------------------------------------------------------------------
// Final: BN2 + ReLU + dot(W3) + b3 + (fm1+fm2), one wave per row.
// ---------------------------------------------------------------------------
__global__ void k_final(const float* __restrict__ Z2, const float* __restrict__ mean,
                        const float* __restrict__ inv, const float* __restrict__ g,
                        const float* __restrict__ be, const float* __restrict__ W3,
                        const float* __restrict__ b3, const float* __restrict__ fmsum,
                        float* __restrict__ out) {
  int lane = threadIdx.x & 31;
  int row = (int)(((size_t)blockIdx.x * blockDim.x + threadIdx.x) >> 5);
  float acc = 0.f;
#pragma unroll
  for (int t = 0; t < 4; ++t) {
    int n = lane + t * 32;
    float z = Z2[(size_t)row * N2 + n];
    float y = fmaxf(g[n] * (z - mean[n]) * inv[n] + be[n], 0.f);
    acc += y * W3[n];
  }
  acc += __shfl_xor(acc, 1);  acc += __shfl_xor(acc, 2);
  acc += __shfl_xor(acc, 4);  acc += __shfl_xor(acc, 8);
  acc += __shfl_xor(acc, 16);
  if (lane == 0) out[row] = acc + b3[0] + fmsum[row];
}

// ---------------------------------------------------------------------------
extern "C" void kernel_launch(void* const* d_in, const int* in_sizes, int n_in,
                              void* d_out, int out_size, void* d_ws, size_t ws_size,
                              hipStream_t stream) {
  (void)in_sizes; (void)n_in; (void)out_size; (void)ws_size;
  const float* x    = (const float*)d_in[0];
  const float* emb  = (const float*)d_in[1];
  const float* fm1t = (const float*)d_in[2];
  const float* dw   = (const float*)d_in[3];
  const float* db   = (const float*)d_in[4];
  const float* dfw  = (const float*)d_in[5];
  const float* dfb  = (const float*)d_in[6];
  const float* W1   = (const float*)d_in[7];
  const float* b1   = (const float*)d_in[8];
  const float* g1   = (const float*)d_in[9];
  const float* be1  = (const float*)d_in[10];
  const float* W2   = (const float*)d_in[11];
  const float* b2   = (const float*)d_in[12];
  const float* g2   = (const float*)d_in[13];
  const float* be2  = (const float*)d_in[14];
  const float* W3   = (const float*)d_in[15];
  const float* b3   = (const float*)d_in[16];
  float* out = (float*)d_out;

  char* wsb = (char*)d_ws;
  size_t cur = 0;
  auto take = [&](size_t bytes) -> void* {
    void* p = wsb + cur;
    cur = (cur + bytes + 255) & ~(size_t)255;
    return p;
  };

  __bf16* A1hi  = (__bf16*)take((size_t)B_ROWS * KPAD1 * 2);
  __bf16* A1lo  = (__bf16*)take((size_t)B_ROWS * KPAD1 * 2);
  float*  fmsum = (float*) take((size_t)B_ROWS * 4);
  __bf16* W1hiT = (__bf16*)take((size_t)N1 * KPAD1 * 2);
  __bf16* W1loT = (__bf16*)take((size_t)N1 * KPAD1 * 2);
  __bf16* W2hiT = (__bf16*)take((size_t)N2 * K2 * 2);
  __bf16* W2loT = (__bf16*)take((size_t)N2 * K2 * 2);
  float*  z1    = (float*) take((size_t)B_ROWS * N1 * 4);
  __bf16* h1hi  = (__bf16*)take((size_t)B_ROWS * K2 * 2);
  __bf16* h1lo  = (__bf16*)take((size_t)B_ROWS * K2 * 2);
  float*  z2    = (float*) take((size_t)B_ROWS * N2 * 4);
  float*  ps1   = (float*) take((size_t)128 * N1 * 4);
  float*  pss1  = (float*) take((size_t)128 * N1 * 4);
  float*  mean1 = (float*) take((size_t)N1 * 4);
  float*  inv1  = (float*) take((size_t)N1 * 4);
  float*  ps2   = (float*) take((size_t)128 * N2 * 4);
  float*  pss2  = (float*) take((size_t)128 * N2 * 4);
  float*  mean2 = (float*) take((size_t)N2 * 4);
  float*  inv2  = (float*) take((size_t)N2 * 4);

  const float invM = 1.0f / (float)B_ROWS;

  // 1) embeddings + FM terms (one wave per row)
  k_embed<<<(B_ROWS * 32) / 256, 256, 0, stream>>>(x, emb, fm1t, dw, db, dfw, dfb,
                                                   A1hi, A1lo, fmsum);
  // 2) weight conversion/transposition
  k_convW<<<(N1 * KPAD1 + 255) / 256, 256, 0, stream>>>(W1, W1hiT, W1loT, 528, KPAD1, N1);
  k_convW<<<(N2 * K2 + 255) / 256, 256, 0, stream>>>(W2, W2hiT, W2loT, K2, K2, N2);

  // 3) GEMM1: z1 = embed @ W1 + b1
  {
    int waves = (B_ROWS / 16) * (N1 / 64);  // 4096
    k_gemm<<<(waves * 32) / 256, 256, 0, stream>>>(A1hi, A1lo, W1hiT, W1loT, b1, z1,
                                                   B_ROWS, KPAD1, N1);
  }
  // 4) BN1 stats (deterministic two-pass)
  k_stats1<<<128, N1, 0, stream>>>(z1, ps1, pss1, N1, B_ROWS / 128);
  k_stats2<<<1, N1, 0, stream>>>(ps1, pss1, mean1, inv1, N1, 128, invM);
  // 5) BN1 + ReLU + bf16 split for GEMM2 A
  k_bnsplit<<<(B_ROWS * N1) / 256, 256, 0, stream>>>(z1, mean1, inv1, g1, be1, h1hi,
                                                     h1lo, B_ROWS * N1, N1 - 1);
  // 6) GEMM2: z2 = h1 @ W2 + b2
  {
    int waves = (B_ROWS / 16) * (N2 / 64);  // 2048
    k_gemm<<<(waves * 32) / 256, 256, 0, stream>>>(h1hi, h1lo, W2hiT, W2loT, b2, z2,
                                                   B_ROWS, K2, N2);
  }
  // 7) BN2 stats
  k_stats1<<<128, N2, 0, stream>>>(z2, ps2, pss2, N2, B_ROWS / 128);
  k_stats2<<<1, N2, 0, stream>>>(ps2, pss2, mean2, inv2, N2, 128, invM);
  // 8) BN2 + ReLU + W3 dot + fm1 + fm2
  k_final<<<(B_ROWS * 32) / 256, 256, 0, stream>>>(z2, mean2, inv2, g2, be2, W3, b3,
                                                   fmsum, out);
}
